// LAWC_38783554683254
// MI455X (gfx1250) — compile-verified, matrix operand order checked by reading, not run on hardware
//
#include <hip/hip_runtime.h>
#include <hip/hip_bf16.h>
#include <math.h>

// Problem constants (match reference)
#define Bsz   8
#define Cch   256
#define Hh    96
#define Ww    96
#define Np    16
#define HWp   (Hh*Ww)      // 9216
#define NTAP  9

typedef __bf16 v8bf  __attribute__((ext_vector_type(8)));
typedef __bf16 v16bf __attribute__((ext_vector_type(16)));
typedef float  v8f   __attribute__((ext_vector_type(8)));

__device__ __forceinline__ float sigmoidf_(float x) { return 1.0f / (1.0f + expf(-x)); }

__device__ __forceinline__ void wait_asynccnt0() {
#if __has_builtin(__builtin_amdgcn_s_wait_asynccnt)
    __builtin_amdgcn_s_wait_asynccnt(0);
#else
    asm volatile("s_wait_asynccnt 0x0" ::: "memory");
#endif
}

// ---------------------------------------------------------------------------
// Kernel 1: fold channel gates into WMMA-swizzled bf16 weight matrices.
//   BwN[tap][cb][lane][e] = prototypes[n][c][tap] * a_c[n][c]   (numerator B)
//   BwS[tap][cb][lane][e] = a_c[n][c]^2                          (sum-sq B)
// with n = lane&15, c = cb*32 + (e&7) + 8*(lane>>4) + 16*(e>>3)
// (B operand layout for V_WMMA_F32_16X16X32_BF16, K tap-major)
// Also scale[n] = softmax(proto_w)[n] / max(||proto_n||, 1e-6).
// BwS is written contiguously after BwN (single base for async staging).
// ---------------------------------------------------------------------------
__global__ void lawc_prep(const float* __restrict__ prot,
                          const float* __restrict__ Wc_w,
                          const float* __restrict__ Wc_b,
                          const float* __restrict__ proto_w,
                          __bf16* __restrict__ BwN,
                          __bf16* __restrict__ BwS,
                          float* __restrict__ scale)
{
    __shared__ float pbar[Np * Cch];
    __shared__ float acg[Np * Cch];
    __shared__ float pn[Np];
    const int t = threadIdx.x;

    // p_bar = mean over 3x3
    for (int i = t; i < Np * Cch; i += 256) {
        const float* p = prot + i * 9;
        float s = 0.f;
        #pragma unroll
        for (int k = 0; k < 9; k++) s += p[k];
        pbar[i] = s * (1.0f / 9.0f);
    }
    __syncthreads();

    // a_c = sigmoid(p_bar @ Wc_w^T + b)
    for (int i = t; i < Np * Cch; i += 256) {
        const int n = i >> 8, c = i & 255;
        const float* w = Wc_w + (size_t)c * Cch;
        const float* x = pbar + n * Cch;
        float s = Wc_b[c];
        for (int k = 0; k < Cch; k++) s += x[k] * w[k];
        acg[i] = sigmoidf_(s);
    }

    // prototype norms
    if (t < Np) {
        const float* p = prot + (size_t)t * Cch * 9;
        float s = 0.f;
        for (int k = 0; k < Cch * 9; k++) s += p[k] * p[k];
        pn[t] = fmaxf(sqrtf(s), 1e-6f);
    }
    __syncthreads();

    // scale[n] = softmax(proto_w)[n] / p_norm[n]
    if (t < Np) {
        float mx = -1e30f;
        for (int n = 0; n < Np; n++) mx = fmaxf(mx, proto_w[n]);
        float den = 0.f;
        for (int n = 0; n < Np; n++) den += expf(proto_w[n] - mx);
        scale[t] = (expf(proto_w[t] - mx) / den) / pn[t];
    }

    // swizzled B operands
    for (int idx = t; idx < NTAP * 8 * 32 * 16; idx += 256) {
        const int e   = idx & 15;
        const int l   = (idx >> 4) & 31;
        const int cb  = (idx >> 9) & 7;
        const int tap = idx >> 12;
        const int c = cb * 32 + (e & 7) + ((l >> 4) << 3) + ((e >> 3) << 4);
        const int n = l & 15;
        const float g = acg[n * Cch + c];
        BwN[idx] = (__bf16)(prot[((size_t)(n * Cch + c)) * 9 + tap] * g);
        BwS[idx] = (__bf16)(g * g);
    }
}

// ---------------------------------------------------------------------------
// Kernel 2: a_s = sigmoid(1x1 conv), Fs = bf16(F * a_s), channels-last layout
// Fs[b][pixel][c].  64 pixels per block, LDS-tiled for coalesced b128 stores.
// ---------------------------------------------------------------------------
__global__ void lawc_gate(const float* __restrict__ F,
                          const float* __restrict__ sp_w,
                          const float* __restrict__ sp_b,
                          __bf16* __restrict__ Fs)
{
    __shared__ float red[256];
    __shared__ float as_l[64];
    __shared__ __align__(16) __bf16 tile[64 * Cch];   // 32 KB

    const int b  = blockIdx.x;
    const int p0 = blockIdx.y * 64;
    const int t  = threadIdx.x;
    const int pl = t & 63;      // pixel within tile
    const int cg = t >> 6;      // channel group 0..3
    const float* Fb = F + (size_t)b * Cch * HWp;

    // pass 1: spatial-gate dot product over channels (coalesced over pixels)
    float acc = 0.f;
    for (int c = cg; c < Cch; c += 4)
        acc += Fb[(size_t)c * HWp + p0 + pl] * sp_w[c];
    red[t] = acc;
    __syncthreads();
    if (t < 64) {
        float s = red[t] + red[t + 64] + red[t + 128] + red[t + 192] + sp_b[0];
        as_l[t] = sigmoidf_(s);
    }
    __syncthreads();

    // pass 2: gate, convert to bf16, transpose to channels-last in LDS
    for (int c = cg; c < Cch; c += 4) {
        float f = Fb[(size_t)c * HWp + p0 + pl];
        tile[pl * Cch + c] = (__bf16)(f * as_l[pl]);
    }
    __syncthreads();

    // coalesced 16B stores: 64*256 bf16 = 2048 x uint4
    const uint4* src = (const uint4*)tile;
    uint4* dst = (uint4*)(Fs + ((size_t)b * HWp + p0) * Cch);
    for (int i = t; i < 2048; i += 256) dst[i] = src[i];
}

// ---------------------------------------------------------------------------
// Async-stage one tap's B operands (BwN slice + BwS slice, 16 KB total) into
// LDS double buffer via CDNA5 GLOBAL_LOAD_ASYNC_TO_LDS_B128 (ASYNCcnt).
// GVS addressing: saddr = Bw base (SGPR pair), vaddr = 32-bit byte offset,
// vdst = LDS byte offset.  1024 x 16B chunks, 4 per thread.
// Global layout: BwN at element 0, BwS at element 36864 (contiguous).
// ---------------------------------------------------------------------------
__device__ __forceinline__ void lawc_stage_tap(const __bf16* __restrict__ Bbase,
                                               __bf16* sB, int buf, int tap, int t)
{
    #pragma unroll
    for (int i = 0; i < 4; i++) {
        const int ch = t + i * 256;                       // 0..1023
        const int matOfs = (ch & 512) ? (NTAP * 4096) : 0; // BwS after BwN
        const unsigned goff = (unsigned)((matOfs + tap * 4096 + (ch & 511) * 8)
                                         * (int)sizeof(__bf16));
        const unsigned loff = (unsigned)(size_t)(sB + buf * 8192 + ch * 8);
        asm volatile("global_load_async_to_lds_b128 %0, %1, %2 offset:0"
                     :: "v"(loff), "v"(goff), "s"(Bbase)
                     : "memory");
    }
}

// ---------------------------------------------------------------------------
// Kernel 3: implicit-GEMM 3x3 conv via WMMA.
// 1 wave = 16 pixels (M) x 16 prototypes (N); K = 9 taps * 256 channels.
// Two accumulators share the A operand: accG uses a*a elementwise.
// B operands are identical for every wave -> double-buffered async staging in
// LDS (16 KB per tap, prefetch tap+1 while computing tap).
// ---------------------------------------------------------------------------
__global__ void lawc_simconv(const __bf16* __restrict__ Fs,
                             const __bf16* __restrict__ Bw,   // BwN, BwS contiguous
                             const float* __restrict__ scale,
                             float* __restrict__ out)
{
    __shared__ __align__(16) __bf16 sB[2 * 8192];   // 2 bufs x (BwN 8KB + BwS 8KB)

    const int wave = (blockIdx.x * blockDim.x + threadIdx.x) >> 5;
    const int lane = threadIdx.x & 31;
    const int t    = threadIdx.x;
    const int b  = wave / (Hh * 6);
    const int rm = wave % (Hh * 6);
    const int y  = rm / 6;
    const int x0 = (rm % 6) * 16;
    const int m    = lane & 15;   // pixel row of A held by this lane
    const int half = lane >> 4;

    v8f accN = {};
    v8f accG = {};

    // prologue: stage tap 0
    lawc_stage_tap(Bw, sB, 0, 0, t);
    wait_asynccnt0();
    __syncthreads();

    for (int tap = 0; tap < NTAP; tap++) {
        const int buf = tap & 1;
        // async-prefetch next tap's B while computing this one
        if (tap + 1 < NTAP) lawc_stage_tap(Bw, sB, buf ^ 1, tap + 1, t);

        const int dy = tap / 3 - 1, dx = tap % 3 - 1;
        const int yy = y + dy, xx = x0 + m + dx;
        const bool valid = (yy >= 0) & (yy < Hh) & (xx >= 0) & (xx < Ww);
        const __bf16* ap   = Fs + ((size_t)b * HWp + (size_t)yy * Ww + xx) * Cch;
        const __bf16* sbuf = sB + buf * 8192;
        #pragma unroll
        for (int cb = 0; cb < 8; cb++) {
            v16bf a = {};
            if (valid) {
                v8bf lo = *(const v8bf*)(ap + cb * 32 + 8 * half);
                v8bf hi = *(const v8bf*)(ap + cb * 32 + 16 + 8 * half);
                a = __builtin_shufflevector(lo, hi,
                        0, 1, 2, 3, 4, 5, 6, 7, 8, 9, 10, 11, 12, 13, 14, 15);
            }
            v16bf bn = *(const v16bf*)(sbuf + cb * 512 + lane * 16);          // LDS
            v16bf bs = *(const v16bf*)(sbuf + 4096 + cb * 512 + lane * 16);   // LDS
            v16bf a2 = a * a;   // shared-A trick: sum-of-squares GEMM operand
            accN = __builtin_amdgcn_wmma_f32_16x16x32_bf16(
                       false, a,  false, bn, (short)0, accN, false, false);
            accG = __builtin_amdgcn_wmma_f32_16x16x32_bf16(
                       false, a2, false, bs, (short)0, accG, false, false);
        }
        // next tap's staging must be complete and visible block-wide
        wait_asynccnt0();
        __syncthreads();
    }

    // Epilogue: val[m,n] = scale[n]*num/sqrt(max(g,1e-6)); reduce over n.
    // D layout: lane holds column n=lane&15; VGPR r holds row m=r+8*half.
    const int n = lane & 15;
    const float sc = scale[n];
    float res[8];
    #pragma unroll
    for (int r = 0; r < 8; r++) {
        float v = sc * accN[r] * rsqrtf(fmaxf(accG[r], 1e-6f));
        v += __shfl_xor(v, 1, 16);
        v += __shfl_xor(v, 2, 16);
        v += __shfl_xor(v, 4, 16);
        v += __shfl_xor(v, 8, 16);
        res[r] = v;
    }
    if (n == 0) {
        float* op = out + (size_t)b * HWp + (size_t)y * Ww + x0 + 8 * half;
        #pragma unroll
        for (int r = 0; r < 8; r++) op[r] = res[r];
    }
}

// ---------------------------------------------------------------------------
extern "C" void kernel_launch(void* const* d_in, const int* in_sizes, int n_in,
                              void* d_out, int out_size, void* d_ws, size_t ws_size,
                              hipStream_t stream)
{
    const float* F       = (const float*)d_in[0];
    const float* prot    = (const float*)d_in[1];
    const float* Wc_w    = (const float*)d_in[2];
    const float* Wc_b    = (const float*)d_in[3];
    const float* sp_w    = (const float*)d_in[4];
    const float* sp_b    = (const float*)d_in[5];
    const float* proto_w = (const float*)d_in[6];
    float* out = (float*)d_out;

    char* ws = (char*)d_ws;
    const size_t fsBytes = (size_t)Bsz * HWp * Cch * sizeof(__bf16);   // ~37.75 MB
    const size_t bwBytes = (size_t)NTAP * 8 * 32 * 16 * sizeof(__bf16); // 73728 B
    __bf16* Fs    = (__bf16*)ws;
    __bf16* BwN   = (__bf16*)(ws + fsBytes);
    __bf16* BwS   = (__bf16*)(ws + fsBytes + bwBytes);   // contiguous after BwN
    float*  scale = (float*)(ws + fsBytes + 2 * bwBytes);

    lawc_prep<<<1, 256, 0, stream>>>(prot, Wc_w, Wc_b, proto_w, BwN, BwS, scale);
    lawc_gate<<<dim3(Bsz, HWp / 64), 256, 0, stream>>>(F, sp_w, sp_b, Fs);
    lawc_simconv<<<(Bsz * Hh * 6) / 8, 256, 0, stream>>>(Fs, BwN, scale, out);
}
